// rnn_17910013624852
// MI455X (gfx1250) — matrix operand order repaired and checked by reading, not compiled, not run on hardware
//
#include <hip/hip_runtime.h>

// CDNA5 (gfx1250) fused Elman RNN: h_t = tanh(W_ih x_t + b_ih + W_hh h_{t-1} + b_hh)
// Transposed GEMM D[o][b]: weights are the WMMA A-matrix (resident in registers
// as f16 fragments), x/h are B-matrices. fp32 accumulate, hardware v_tanh_f32.

#define S_LEN 1024
#define BATCH 4096
#define NIN   64
#define NHID  128

typedef __attribute__((ext_vector_type(16))) _Float16 v16h;
typedef __attribute__((ext_vector_type(2)))  _Float16 v2h;
typedef __attribute__((ext_vector_type(8)))  float    v8f;

union H16x16 {
  v16h v;
  unsigned int u[8];
  uint4 q[2];
};
union F32x8 {
  v8f v;
  float f[8];
  float4 q[2];
};

// Pack two fp32 -> packed f16 pair in one v_cvt_pk_f16_f32 (RTZ).
__device__ __forceinline__ unsigned int pk2(float a, float b) {
  return __builtin_bit_cast(unsigned int, __builtin_amdgcn_cvt_pkrtz(a, b));
}

__device__ __forceinline__ float fast_tanh(float x) {
#if __has_builtin(__builtin_amdgcn_tanhf)
  return __builtin_amdgcn_tanhf(x);           // v_tanh_f32 (gfx1250 TRANS op)
#elif __has_builtin(__builtin_amdgcn_exp2f) && __has_builtin(__builtin_amdgcn_rcpf)
  // tanh(x) = 1 - 2/(1 + e^{2x}); branch-free, saturates correctly at +-inf.
  float e = __builtin_amdgcn_exp2f(x * 2.885390081777927f);  // e^{2x}
  float r = __builtin_amdgcn_rcpf(e + 1.0f);
  return 1.0f - (r + r);
#else
  return tanhf(x);
#endif
}

__device__ __forceinline__ v8f wmma_f16(const H16x16& a, const H16x16& b, v8f c) {
  return __builtin_amdgcn_wmma_f32_16x16x32_f16(false, a.v, false, b.v,
                                                (short)0, c, false, false);
}

__global__ __launch_bounds__(128) void rnn_elman_gfx1250(
    const float* __restrict__ x,     // [S, B, 64]
    const float* __restrict__ W_ih,  // [128, 64]
    const float* __restrict__ b_ih,  // [128]
    const float* __restrict__ W_hh,  // [128, 128]
    const float* __restrict__ b_hh,  // [128]
    float* __restrict__ out)         // [B, 128]
{
  // Double-buffered hidden state tile, f16, [buf][batch-row][hidden]: 8 KB.
  __shared__ __align__(16) unsigned short hbuf[2][16][NHID];

  const int tid  = threadIdx.x;
  const int lane = tid & 31;
  const int wave = tid >> 5;    // 0..3: owns outputs [wave*32, wave*32+32)
  const int lrow = lane & 15;   // batch row within tile / weight output row
  const int hi   = lane >> 4;   // half-wave selector for K/M sub-chunks
  const int b0   = blockIdx.x * 16;
  const int obase = wave * 32;

  // Zero h0 buffer.
  {
    unsigned int* z = reinterpret_cast<unsigned int*>(&hbuf[0][0][0]);
    for (int i = tid; i < 16 * NHID / 2; i += 128) z[i] = 0u;
  }

  // ---- Weight A-fragments into registers (persist whole kernel) ----
  // A k order used here (contiguous, B matches): lanes 0-15 -> row M=l, K [kt*32+0..15];
  // lanes 16-31 -> row M=l-16, K [kt*32+16..31].
  H16x16 a_ih[2][2];  // [mt][kt], K = 0..63
  H16x16 a_hh[2][4];  // [mt][kt], K = 0..127
#pragma unroll
  for (int mt = 0; mt < 2; ++mt) {
    const int o = obase + mt * 16 + lrow;
#pragma unroll
    for (int kt = 0; kt < 2; ++kt) {
      const float* p = W_ih + o * NIN + kt * 32 + hi * 16;
#pragma unroll
      for (int e = 0; e < 4; ++e) {
        a_ih[mt][kt].u[e]     = pk2(p[2 * e],      p[2 * e + 1]);
        a_ih[mt][kt].u[4 + e] = pk2(p[8 + 2 * e], p[8 + 2 * e + 1]);
      }
    }
#pragma unroll
    for (int kt = 0; kt < 4; ++kt) {
      const float* p = W_hh + o * NHID + kt * 32 + hi * 16;
#pragma unroll
      for (int e = 0; e < 4; ++e) {
        a_hh[mt][kt].u[e]     = pk2(p[2 * e],      p[2 * e + 1]);
        a_hh[mt][kt].u[4 + e] = pk2(p[8 + 2 * e], p[8 + 2 * e + 1]);
      }
    }
  }

  // ---- Bias fragments in C/D layout: elem r -> o = base + hi*8 + r ----
  F32x8 bias[2];
#pragma unroll
  for (int mt = 0; mt < 2; ++mt) {
    const int o = obase + mt * 16 + hi * 8;
#pragma unroll
    for (int e = 0; e < 8; ++e) bias[mt].f[e] = b_ih[o + e] + b_hh[o + e];
  }

  __syncthreads();

  F32x8 h[2];  // current post-tanh hidden fragments (fp32)
  h[0].v = v8f{};
  h[1].v = v8f{};

  for (int s = 0; s < S_LEN; ++s) {
    const int cur = s & 1;
    const int nxt = cur ^ 1;

    // h B-fragments: lane -> batch row lrow, K = kt*32 + hi*16 .. +15 (32B).
    H16x16 bh[4];
#pragma unroll
    for (int kt = 0; kt < 4; ++kt) {
      const uint4* p = reinterpret_cast<const uint4*>(&hbuf[cur][lrow][kt * 32 + hi * 16]);
      bh[kt].q[0] = p[0];
      bh[kt].q[1] = p[1];
    }

    // x B-fragments from global: 16 contiguous fp32 per lane per kt (64B),
    // packed to f16 with v_cvt_pk_f16_f32 (one instruction per u32 word).
    const float* xrow = x + ((size_t)s * BATCH + (size_t)(b0 + lrow)) * NIN;
    H16x16 bx[2];
#pragma unroll
    for (int kt = 0; kt < 2; ++kt) {
      const float4* p4 = reinterpret_cast<const float4*>(xrow + kt * 32 + hi * 16);
      float4 va = p4[0], vb = p4[1], vc = p4[2], vd = p4[3];
      bx[kt].u[0] = pk2(va.x, va.y);
      bx[kt].u[1] = pk2(va.z, va.w);
      bx[kt].u[2] = pk2(vb.x, vb.y);
      bx[kt].u[3] = pk2(vb.z, vb.w);
      bx[kt].u[4] = pk2(vc.x, vc.y);
      bx[kt].u[5] = pk2(vc.z, vc.w);
      bx[kt].u[6] = pk2(vd.x, vd.y);
      bx[kt].u[7] = pk2(vd.z, vd.w);
    }
    if (s + 1 < S_LEN) {
      __builtin_prefetch(xrow + (size_t)BATCH * NIN, 0, 1);  // global_prefetch_b8
    }

#pragma unroll
    for (int mt = 0; mt < 2; ++mt) {
      v8f acc = bias[mt].v;
      acc = wmma_f16(a_ih[mt][0], bx[0], acc);
      acc = wmma_f16(a_ih[mt][1], bx[1], acc);
#pragma unroll
      for (int kt = 0; kt < 4; ++kt) acc = wmma_f16(a_hh[mt][kt], bh[kt], acc);

      F32x8 t;
      t.v = acc;
#pragma unroll
      for (int e = 0; e < 8; ++e) t.f[e] = fast_tanh(t.f[e]);
      h[mt] = t;

      // Pack 8 f16 (contiguous o within this lane) -> one ds_store_b128.
      uint4 pkv;
      pkv.x = pk2(t.f[0], t.f[1]);
      pkv.y = pk2(t.f[2], t.f[3]);
      pkv.z = pk2(t.f[4], t.f[5]);
      pkv.w = pk2(t.f[6], t.f[7]);
      *reinterpret_cast<uint4*>(&hbuf[nxt][lrow][obase + mt * 16 + hi * 8]) = pkv;
    }

    __syncthreads();
  }

  // Final hidden state: out[b][o], contiguous o within lane -> 2x b128 stores.
#pragma unroll
  for (int mt = 0; mt < 2; ++mt) {
    float* po = out + (size_t)(b0 + lrow) * NHID + obase + mt * 16 + hi * 8;
    *reinterpret_cast<float4*>(po)     = h[mt].q[0];
    *reinterpret_cast<float4*>(po + 4) = h[mt].q[1];
  }
}

extern "C" void kernel_launch(void* const* d_in, const int* in_sizes, int n_in,
                              void* d_out, int out_size, void* d_ws, size_t ws_size,
                              hipStream_t stream) {
  (void)in_sizes; (void)n_in; (void)out_size; (void)d_ws; (void)ws_size;
  const float* x    = (const float*)d_in[0];
  const float* W_ih = (const float*)d_in[1];
  const float* b_ih = (const float*)d_in[2];
  const float* W_hh = (const float*)d_in[3];
  const float* b_hh = (const float*)d_in[4];
  float* out = (float*)d_out;

  dim3 grid(BATCH / 16);  // 256 workgroups, one 16-row batch tile each
  dim3 block(128);        // 4 waves, each owns 32 of the 128 hidden outputs
  rnn_elman_gfx1250<<<grid, block, 0, stream>>>(x, W_ih, b_ih, W_hh, b_hh, out);
}
